// SparseTopKTokenMixer_37048387895874
// MI455X (gfx1250) — matrix operand order repaired and checked by reading, not compiled, not run on hardware
//
#include <hip/hip_runtime.h>
#include <hip/hip_bf16.h>

#define LN_EPS   1e-5f
#define NORM_EPS 1e-12f
#define TOPK 8
#define B_ 32
#define N_ 1025
#define T_ 1024
#define D_ 384
#define KSTEPS (D_ / 32)   // 12 WMMA K-steps

typedef __attribute__((ext_vector_type(16))) _Float16 v16h;
typedef __attribute__((ext_vector_type(8)))  _Float16 v8h;
typedef __attribute__((ext_vector_type(8)))  float    v8f;

// 6 K-step (half-tile) B-fragment load: 12x global_load_b128 per wave.
#define LOAD6(buf, baseptr)                                                  \
    _Pragma("unroll") for (int _k = 0; _k < 6; ++_k)                         \
        (buf)[_k] = *(const v16h*)((baseptr) + _k * 32);

// 6 dependent WMMA accumulation steps on one buffer.
#define MMA6(cacc, aoff, buf)                                                \
    _Pragma("unroll") for (int _k = 0; _k < 6; ++_k)                         \
        (cacc) = __builtin_amdgcn_wmma_f32_16x16x32_f16(                     \
            false, afrag[(aoff) + _k], false, (buf)[_k], (short)0, (cacc),   \
            false, false);

// ---------------------------------------------------------------------------
// Kernel 1: LayerNorm + L2-normalized token_repr (f16) + per-batch gate.
// One 128-thread block per (b, n) row; each thread owns 3 of the 384 elems.
// ---------------------------------------------------------------------------
__global__ void ln_kernel(const float* __restrict__ x,
                          const float* __restrict__ ln_gamma,
                          const float* __restrict__ ln_beta,
                          const float* __restrict__ gate_w,
                          const float* __restrict__ gate_b,
                          float* __restrict__ out,
                          float* __restrict__ gate,      // [B]
                          float* __restrict__ hidden,    // [B,N,D] f32
                          _Float16* __restrict__ repr)   // [B,T,D] f16 (normalized)
{
    __shared__ float red[128];
    const int row = blockIdx.x;            // 0 .. B*N-1
    const int b   = row / N_;
    const int n   = row % N_;
    const int tid = threadIdx.x;
    const float* xr = x + (size_t)row * D_;

    const float v0 = xr[tid];
    const float v1 = xr[tid + 128];
    const float v2 = xr[tid + 256];

    // mean
    red[tid] = v0 + v1 + v2;
    __syncthreads();
    for (int off = 64; off; off >>= 1) {
        if (tid < off) red[tid] += red[tid + off];
        __syncthreads();
    }
    const float mean = red[0] * (1.0f / D_);
    __syncthreads();

    // E[x^2] -> var
    red[tid] = v0 * v0 + v1 * v1 + v2 * v2;
    __syncthreads();
    for (int off = 64; off; off >>= 1) {
        if (tid < off) red[tid] += red[tid + off];
        __syncthreads();
    }
    const float var = red[0] * (1.0f / D_) - mean * mean;
    __syncthreads();

    const float rstd = rsqrtf(var + LN_EPS);
    const float h0 = (v0 - mean) * rstd * ln_gamma[tid]       + ln_beta[tid];
    const float h1 = (v1 - mean) * rstd * ln_gamma[tid + 128] + ln_beta[tid + 128];
    const float h2 = (v2 - mean) * rstd * ln_gamma[tid + 256] + ln_beta[tid + 256];

    float* hr = hidden + (size_t)row * D_;
    hr[tid] = h0; hr[tid + 128] = h1; hr[tid + 256] = h2;

    if (n == 0) {
        // per-batch gate = sigmoid(cls_hidden @ gate_w + gate_b)
        red[tid] = h0 * gate_w[tid] + h1 * gate_w[tid + 128] + h2 * gate_w[tid + 256];
        __syncthreads();
        for (int off = 64; off; off >>= 1) {
            if (tid < off) red[tid] += red[tid + off];
            __syncthreads();
        }
        if (tid == 0) {
            const float logit = red[0] + gate_b[0];
            gate[b] = 1.0f / (1.0f + __expf(-logit));
        }
        // delta for CLS row is zero -> out = x
        float* orow = out + (size_t)row * D_;
        orow[tid] = v0; orow[tid + 128] = v1; orow[tid + 256] = v2;
    } else {
        // token_repr = hidden / max(||hidden||, eps), stored as f16
        red[tid] = h0 * h0 + h1 * h1 + h2 * h2;
        __syncthreads();
        for (int off = 64; off; off >>= 1) {
            if (tid < off) red[tid] += red[tid + off];
            __syncthreads();
        }
        const float nrm = sqrtf(red[0]);
        const float inv = 1.0f / fmaxf(nrm, NORM_EPS);
        _Float16* rr = repr + ((size_t)b * T_ + (n - 1)) * D_;
        rr[tid]       = (_Float16)(h0 * inv);
        rr[tid + 128] = (_Float16)(h1 * inv);
        rr[tid + 256] = (_Float16)(h2 * inv);
    }
}

// ---------------------------------------------------------------------------
// Kernel 2: WMMA affinity strip (16 rows x 1024 cols) -> top-8 softmax -> mix.
// Block = 128 threads (4 waves). Wave w owns column tiles [w*16, w*16+15].
// B fragments double-buffered in half-tile (6 K-step) chunks; A fragments
// (96 VGPRs) stay register-resident.  launch_bounds(128,1) +
// amdgpu_waves_per_eu(1) lift the VGPR cap so nothing spills (working set
// ~220 VGPRs; occupancy is LDS-bound anyway at ~4 blocks/WGP).
// ---------------------------------------------------------------------------
__global__ void
__launch_bounds__(128, 1)
__attribute__((amdgpu_waves_per_eu(1)))
mix_kernel(const float* __restrict__ x,
           const float* __restrict__ layer_scale,  // [D]
           const float* __restrict__ gate,         // [B]
           const float* __restrict__ hidden,       // [B,N,D]
           const _Float16* __restrict__ repr,      // [B,T,D]
           float* __restrict__ out)
{
    extern __shared__ char smem[];
    float* aff   = (float*)smem;                   // [16][1024]  64 KB
    float* candv = aff + 16 * T_;                  // [16][64]     4 KB
    int*   candi = (int*)(candv + 16 * 64);        // [16][64]     4 KB
    float* wts   = (float*)(candi + 16 * 64);      // [16][8]
    int*   idxs  = (int*)(wts + 16 * TOPK);        // [16][8]

    const int tile_m = blockIdx.x;             // 0..63 (16-row tiles)
    const int b      = blockIdx.y;             // 0..31
    const int tid    = threadIdx.x;
    const int lane   = tid & 31;
    const int wave   = tid >> 5;               // 0..3
    const int half   = lane >> 4;              // 0 or 1
    const int lrow   = lane & 15;

    const _Float16* reprB = repr + (size_t)b * T_ * D_;

    // --- Preload A fragments for the 16 rows of this tile (all 12 K-steps).
    // A 16x32 f16 order used here (contiguous k, not AMD's table order): lanes 0-15 hold row=lane, K {0..15};
    // lanes 16-31 hold row=lane-16, K {16..31} (matches B fragment order).
    const _Float16* arow = reprB + (size_t)(tile_m * 16 + lrow) * D_;
    v16h afrag[KSTEPS];
#pragma unroll
    for (int ks = 0; ks < KSTEPS; ++ks) {
        const int k0 = ks * 32;
        const v8h lo = *(const v8h*)(arow + k0 + half * 16);
        const v8h hi = *(const v8h*)(arow + k0 + half * 16 + 8);
        v16h a;
#pragma unroll
        for (int i = 0; i < 8; ++i) { a[i] = lo[i]; a[8 + i] = hi[i]; }
        afrag[ks] = a;
    }

    // --- Affinity strip: 16 column tiles per wave, software-pipelined B.
    // B 32x16 order used here (contiguous k, same as A): lanes 0-15 col=lane hold K 0..15 (2 f16 per VGPR);
    // lanes 16-31 col=lane-16 hold K 16..31.  B(k,n) = R[n][k] => read 16
    // consecutive f16 from the column's repr row.
    const _Float16* colbase = reprB + (size_t)(wave * 256 + lrow) * D_ + half * 16;

    v16h bufA[6], bufB[6];
    LOAD6(bufA, colbase);                          // (t=0, half 0)
    v8f c = {};
    for (int t = 0; t < 15; ++t) {
        const _Float16* tb = colbase + (size_t)t * 16 * D_;
        LOAD6(bufB, tb + 6 * 32);                  // (t, half 1) in flight
        MMA6(c, 0, bufA);                          // consume (t, half 0)
        LOAD6(bufA, tb + 16 * D_);                 // (t+1, half 0) in flight
        MMA6(c, 6, bufB);                          // consume (t, half 1)
        // C layout: vgpr j holds row = half*8 + j, col = lrow.
#pragma unroll
        for (int j = 0; j < 8; ++j) {
            aff[(half * 8 + j) * T_ + (wave * 16 + t) * 16 + lrow] = c[j];
        }
        c = (v8f){};
    }
    {   // epilogue: t = 15 (no next-tile prefetch)
        const _Float16* tb = colbase + (size_t)15 * 16 * D_;
        LOAD6(bufB, tb + 6 * 32);
        MMA6(c, 0, bufA);
        MMA6(c, 6, bufB);
#pragma unroll
        for (int j = 0; j < 8; ++j) {
            aff[(half * 8 + j) * T_ + (wave * 16 + 15) * 16 + lrow] = c[j];
        }
    }
    __syncthreads();

    // --- Parallel top-8: 8 threads per row, 128 columns each.
    {
        const int r   = tid >> 3;                  // 0..15
        const int seg = tid & 7;                   // 0..7
        const float* ar = aff + r * T_ + seg * 128;
        float tv[TOPK]; int ti[TOPK];
#pragma unroll
        for (int j = 0; j < TOPK; ++j) { tv[j] = -3.0e38f; ti[j] = 0; }
        for (int s = 0; s < 128; ++s) {
            const float a = ar[s];
            if (a > tv[TOPK - 1]) {
                int p = TOPK - 1;
                while (p > 0 && tv[p - 1] < a) {
                    tv[p] = tv[p - 1]; ti[p] = ti[p - 1]; --p;
                }
                tv[p] = a; ti[p] = seg * 128 + s;
            }
        }
#pragma unroll
        for (int j = 0; j < TOPK; ++j) {
            candv[r * 64 + seg * TOPK + j] = tv[j];
            candi[r * 64 + seg * TOPK + j] = ti[j];
        }
    }
    __syncthreads();

    // --- Merge 64 candidates/row -> top-8, softmax (1 thread per row).
    if (tid < 16) {
        float tv[TOPK]; int ti[TOPK];
#pragma unroll
        for (int j = 0; j < TOPK; ++j) { tv[j] = -3.0e38f; ti[j] = 0; }
        for (int m = 0; m < 64; ++m) {
            const float a = candv[tid * 64 + m];
            const int   i = candi[tid * 64 + m];
            if (a > tv[TOPK - 1]) {
                int p = TOPK - 1;
                while (p > 0 && tv[p - 1] < a) {
                    tv[p] = tv[p - 1]; ti[p] = ti[p - 1]; --p;
                }
                tv[p] = a; ti[p] = i;
            }
        }
        const float mx = tv[0];
        float e[TOPK];
        float sum = 0.f;
#pragma unroll
        for (int j = 0; j < TOPK; ++j) { e[j] = __expf(tv[j] - mx); sum += e[j]; }
        const float inv = 1.0f / sum;
#pragma unroll
        for (int j = 0; j < TOPK; ++j) {
            wts[tid * TOPK + j]  = e[j] * inv;
            idxs[tid * TOPK + j] = ti[j];
        }
    }
    __syncthreads();

    // --- Sparse mix + layer_scale + gate + residual.
    const float g = gate[b];
    const float* hidB = hidden + (size_t)b * N_ * D_;
    for (int e = tid; e < 16 * D_; e += 128) {
        const int r = e / D_;
        const int d = e - r * D_;
        const int grow = tile_m * 16 + r;      // token index 0..1023
        float acc = 0.f;
#pragma unroll
        for (int j = 0; j < TOPK; ++j) {
            acc += wts[r * TOPK + j] * hidB[(size_t)(1 + idxs[r * TOPK + j]) * D_ + d];
        }
        const float th = hidB[(size_t)(1 + grow) * D_ + d];
        const float delta = g * (acc - th) * layer_scale[d];
        const size_t oidx = ((size_t)b * N_ + 1 + grow) * D_ + d;
        out[oidx] = x[oidx] + delta;
    }
}

// ---------------------------------------------------------------------------
extern "C" void kernel_launch(void* const* d_in, const int* in_sizes, int n_in,
                              void* d_out, int out_size, void* d_ws, size_t ws_size,
                              hipStream_t stream) {
    const float* x           = (const float*)d_in[0];
    const float* ln_gamma    = (const float*)d_in[1];
    const float* ln_beta     = (const float*)d_in[2];
    const float* gate_w      = (const float*)d_in[3];
    const float* gate_b      = (const float*)d_in[4];
    const float* layer_scale = (const float*)d_in[5];
    float* out = (float*)d_out;

    char* ws = (char*)d_ws;
    float*    gate   = (float*)ws;                                   // 32 f32
    float*    hidden = (float*)(ws + 256);                           // B*N*D f32
    _Float16* repr   = (_Float16*)(ws + 256 + (size_t)B_ * N_ * D_ * sizeof(float));

    ln_kernel<<<B_ * N_, 128, 0, stream>>>(x, ln_gamma, ln_beta, gate_w, gate_b,
                                           out, gate, hidden, repr);

    const size_t shmem = (size_t)16 * T_ * sizeof(float)   // aff strip
                       + (size_t)16 * 64 * 8               // candidates
                       + (size_t)16 * TOPK * 8;            // weights + indices
    mix_kernel<<<dim3(T_ / 16, B_), 128, shmem, stream>>>(x, layer_scale, gate,
                                                          hidden, repr, out);
}
